// JointNet_31602369364341
// MI455X (gfx1250) — matrix operand order repaired and checked by reading, not compiled, not run on hardware
//
#include <hip/hip_runtime.h>

// ---------------------------------------------------------------------------
// RNN-T JointNet for MI455X (gfx1250, wave32, WMMA + TDM).
//   Phase 1 : enc_ln[B,T,J] = LN(encoder_out @ W_enc + b_enc)   (f32, small)
//             dec_ln[B,U,J] = LN(decoder_out @ W_dec + b_dec)
//   Phase 1b: Wt[V,J] = bf16(W_out^T)   (one-shot convert+transpose)
//   Phase 2 : out[B,T,U,V] = relu(enc_ln[t] + dec_ln[u]) @ W_out + b_out
//             fused bf16 WMMA GEMM; joint never materialized; B tiles fed
//             by the Tensor Data Mover (tensor_load_to_lds) when available.
// ---------------------------------------------------------------------------

#define Bb 4
#define Tt 512
#define Uu 64
#define Jj 512
#define Vv 1024

typedef __attribute__((ext_vector_type(16))) __bf16        v16bf;
typedef __attribute__((ext_vector_type(8)))  float         v8f;
typedef __attribute__((ext_vector_type(4)))  float         f32x4;
typedef __attribute__((ext_vector_type(2)))  float         f32x2;
typedef __attribute__((ext_vector_type(4)))  unsigned int  u32x4;
typedef __attribute__((ext_vector_type(8)))  int           i32x8;
typedef __attribute__((ext_vector_type(4)))  int           i32x4;

union FragAB { v16bf v; u32x4 q[2]; };

#if defined(__has_builtin)
#if __has_builtin(__builtin_amdgcn_tensor_load_to_lds) && __has_builtin(__builtin_amdgcn_s_wait_tensorcnt)
#define USE_TDM 1
#endif
#endif
#ifndef USE_TDM
#define USE_TDM 0
#endif

__device__ __forceinline__ unsigned int pack_bf16x2(float x, float y) {
    unsigned int ux = __float_as_uint(x), uy = __float_as_uint(y);
    ux = (ux + 0x7FFFu + ((ux >> 16) & 1u)) >> 16;   // RNE truncate to bf16
    uy = (uy + 0x7FFFu + ((uy >> 16) & 1u)) >> 16;
    return (ux & 0xFFFFu) | (uy << 16);
}

__device__ __forceinline__ unsigned short f32_to_bf16(float x) {
    unsigned int ux = __float_as_uint(x);
    return (unsigned short)((ux + 0x7FFFu + ((ux >> 16) & 1u)) >> 16);
}

// ---------------------------------------------------------------------------
// Phase 1: one block per output row. Y[row,:] = LN(X[row,:] @ W + b) * g + be
// ---------------------------------------------------------------------------
__global__ __launch_bounds__(256)
void proj_ln_kernel(const float* __restrict__ X, const float* __restrict__ W,
                    const float* __restrict__ bias, const float* __restrict__ gamma,
                    const float* __restrict__ beta, float* __restrict__ Y)
{
    __shared__ float xs[512];
    __shared__ float redS[256];
    __shared__ float redQ[256];

    const int row = blockIdx.x;
    const int tid = threadIdx.x;

    xs[tid]       = X[(size_t)row * 512 + tid];
    xs[tid + 256] = X[(size_t)row * 512 + tid + 256];
    __syncthreads();

    const int j0 = 2 * tid;
    float a0 = bias[j0], a1 = bias[j0 + 1];

    const f32x2* wp2 = (const f32x2*)W + tid;     // W[e*512 + 2*tid .. +1]
    #pragma unroll 8
    for (int e = 0; e < 512; ++e) {
        f32x2 w = wp2[e * 256];
        float xv = xs[e];
        a0 = __builtin_fmaf(xv, w.x, a0);
        a1 = __builtin_fmaf(xv, w.y, a1);
    }

    redS[tid] = a0 + a1;
    redQ[tid] = a0 * a0 + a1 * a1;
    __syncthreads();
    #pragma unroll
    for (int off = 128; off > 0; off >>= 1) {
        if (tid < off) { redS[tid] += redS[tid + off]; redQ[tid] += redQ[tid + off]; }
        __syncthreads();
    }
    const float mu   = redS[0] * (1.0f / 512.0f);
    const float var  = redQ[0] * (1.0f / 512.0f) - mu * mu;
    const float rstd = rsqrtf(var + 1e-5f);

    Y[(size_t)row * 512 + j0]     = (a0 - mu) * rstd * gamma[j0]     + beta[j0];
    Y[(size_t)row * 512 + j0 + 1] = (a1 - mu) * rstd * gamma[j0 + 1] + beta[j0 + 1];
}

// ---------------------------------------------------------------------------
// Phase 1b: Wt[v][j] = bf16(W_out[j][v]).  64x64 LDS tile transpose.
// grid = (V/64, J/64) = (16, 8), 256 threads.
// ---------------------------------------------------------------------------
__global__ __launch_bounds__(256)
void wt_transpose_kernel(const float* __restrict__ W, unsigned short* __restrict__ Wt)
{
    __shared__ unsigned short tile[64 * 72];   // [v][j], padded pitch

    const int vb  = blockIdx.x * 64;
    const int jb  = blockIdx.y * 64;
    const int tid = threadIdx.x;

    const int tv  = tid & 63;          // coalesced over v on the read side
    const int tj4 = tid >> 6;          // 0..3 -> 16 j rows each
    #pragma unroll
    for (int jj = 0; jj < 16; ++jj) {
        int j = tj4 * 16 + jj;
        float w = W[(size_t)(jb + j) * Vv + vb + tv];
        tile[tv * 72 + j] = f32_to_bf16(w);
    }
    __syncthreads();

    const int wv = tid >> 2;           // 0..63
    const int wj = (tid & 3) * 16;     // 16 shorts = 32B per thread
    const u32x4* src = (const u32x4*)&tile[wv * 72 + wj];
    u32x4* dst = (u32x4*)(Wt + (size_t)(vb + wv) * Jj + jb + wj);
    dst[0] = src[0];
    dst[1] = src[1];
}

// ---------------------------------------------------------------------------
// Phase 2: fused joint + vocab projection via V_WMMA_F32_16X16X32_BF16.
// Block tile 128(M) x 128(N); 8 waves as 4(M) x 2(N); each wave 2x4 C tiles.
// K looped in chunks of 64; A built on the fly in LDS; B (bf16, [n][k]) DMA'd
// by the TDM with LDS padding reproducing the 72-short conflict-free pitch.
// ---------------------------------------------------------------------------
__global__ __launch_bounds__(256)
void joint_wmma_kernel(const float* __restrict__ enc_ln,
                       const float* __restrict__ dec_ln,
                       const unsigned short* __restrict__ Wt,   // [V][J] bf16
                       const float* __restrict__ b_out,
                       float* __restrict__ out)
{
    // row pitch 72 shorts (144B = 36 dwords): conflict-free b128 frag reads
    __shared__ unsigned short Ash[128 * 72];   // A tile, row-major [m][k]
    __shared__ unsigned short Bsh[128 * 72];   // B tile, transposed [n][k]

    const int tid   = threadIdx.x;
    const int lane  = tid & 31;
    const int wave  = tid >> 5;
    const int waveM = wave >> 1;        // 0..3 -> 32 rows each
    const int waveN = wave & 1;         // 0..1 -> 64 cols each
    const int lhalf = lane >> 4;        // 0 | 1
    const int lmod  = lane & 15;

    const int nblk  = blockIdx.x;       // V / 128  = 8
    const int mblk  = blockIdx.y;       // T*U/128  = 256
    const int bat   = blockIdx.z;       // B        = 4
    const int nbase = nblk * 128;
    const int t0    = mblk * 2;         // 128 M-rows = 2 t-rows x 64 u-rows

    // A-fill mapping: 2 threads per m-row, each covers 32 k
    const int frow  = tid >> 1;         // 0..127
    const int fhalf = tid & 1;          // k offset 0 | 32
    const int ft    = frow >> 6;        // 0 | 1
    const int fu    = frow & 63;
    const float* encp = enc_ln + (size_t)(bat * Tt + t0 + ft) * Jj;
    const float* decp = dec_ln + (size_t)(bat * Uu + fu) * Jj;

#if USE_TDM
    // Each wave DMAs its own 16(n) x 64(k) bf16 slice of the B tile.
    const int wsg = __builtin_amdgcn_readfirstlane(wave);   // SGPR-resident
    const unsigned lds_b =
        (unsigned)(size_t)(&Bsh[0]) + (unsigned)(wsg * 16 * 144);
    const unsigned long long gbase =
        (unsigned long long)(size_t)(Wt + (size_t)(nbase + wsg * 16) * Jj);
    // Group1 is chunk-invariant:
    //  data_size=2B | pad_enable | pad_interval=4 (32 dw rows) | pad_amount=3 (4 dw)
    const int g1w0 = (1 << 16) | (1 << 20) | (4 << 22) | (3 << 25);
    const i32x8 g1 = { g1w0,
                       (int)(512u  << 16),   // tensor_dim0 = 512 (k)
                       (int)(1024u << 16),   // tensor_dim1 = 1024 (v rows)
                       (int)(64u   << 16),   // tile_dim0   = 64
                       16,                   // tile_dim1   = 16
                       512,                  // tensor_dim0_stride = 512
                       0, 0 };
    const i32x4 zg = { 0, 0, 0, 0 };
#endif

    v8f acc[2][4];
    #pragma unroll
    for (int mi = 0; mi < 2; ++mi)
        #pragma unroll
        for (int ni = 0; ni < 4; ++ni)
            acc[mi][ni] = v8f{};

    for (int kc = 0; kc < 8; ++kc) {
        const int kbase = kc * 64;

#if USE_TDM
        // ---- B tile: TDM DMA, one descriptor per wave --------------------
        {
            unsigned long long ga = gbase + (unsigned long long)(kbase * 2);
            u32x4 g0;
            g0.x = 1u;                                   // count = 1 (valid)
            g0.y = lds_b;                                // lds_addr
            g0.z = (unsigned)ga;                         // global_addr[31:0]
            g0.w = (unsigned)(ga >> 32) | (2u << 30);    // ga[56:32] | type=2
#if __clang_major__ >= 23
            const i32x8 zg8 = { 0, 0, 0, 0, 0, 0, 0, 0 };
            __builtin_amdgcn_tensor_load_to_lds(g0, g1, zg, zg, zg8, 0);
#else
            __builtin_amdgcn_tensor_load_to_lds(g0, g1, zg, zg, 0);
#endif
        }
#else
        // ---- B tile fallback: b128 copies from bf16 Wt -------------------
        {
            const int n  = tid >> 1;              // 0..127
            const int kh = (tid & 1) * 32;        // 0 | 32
            const u32x4* src =
                (const u32x4*)(Wt + (size_t)(nbase + n) * Jj + kbase + kh);
            unsigned short* dst = &Bsh[n * 72 + kh];
            #pragma unroll
            for (int q = 0; q < 4; ++q)
                *(u32x4*)(dst + q * 8) = src[q];
        }
#endif

        // ---- A tile: bf16(relu(enc[t][k] + dec[u][k])) -------------------
        {
            const int kof = kbase + fhalf * 32;
            const f32x4* ep = (const f32x4*)(encp + kof);
            const f32x4* dp = (const f32x4*)(decp + kof);
            unsigned short* arow = &Ash[frow * 72 + fhalf * 32];
            #pragma unroll
            for (int k4 = 0; k4 < 8; ++k4) {
                f32x4 e4 = ep[k4];
                f32x4 d4 = dp[k4];
                float x0 = fmaxf(e4.x + d4.x, 0.0f);
                float x1 = fmaxf(e4.y + d4.y, 0.0f);
                float x2 = fmaxf(e4.z + d4.z, 0.0f);
                float x3 = fmaxf(e4.w + d4.w, 0.0f);
                *(unsigned int*)(arow + k4 * 4)     = pack_bf16x2(x0, x1);
                *(unsigned int*)(arow + k4 * 4 + 2) = pack_bf16x2(x2, x3);
            }
        }

#if USE_TDM
        __builtin_amdgcn_s_wait_tensorcnt(0);   // each wave waits for its DMA
#endif
        __syncthreads();

        // ---- 2 k-steps of 32, 8 WMMAs each ------------------------------
        #pragma unroll
        for (int ks = 0; ks < 2; ++ks) {
            FragAB a[2], bb[4];
            #pragma unroll
            for (int mi = 0; mi < 2; ++mi) {
                int row = waveM * 32 + mi * 16 + lmod;
                const unsigned short* pa = &Ash[row * 72 + ks * 32 + lhalf * 8];
                a[mi].q[0] = *(const u32x4*)(pa);        // K 0..7   (or 8..15)
                a[mi].q[1] = *(const u32x4*)(pa + 16);   // K 16..23 (or 24..31)
            }
            #pragma unroll
            for (int ni = 0; ni < 4; ++ni) {
                int n = waveN * 64 + ni * 16 + lmod;
                const unsigned short* pb = &Bsh[n * 72 + ks * 32 + lhalf * 8];
                bb[ni].q[0] = *(const u32x4*)(pb);       // K 0..7   (or 8..15)
                bb[ni].q[1] = *(const u32x4*)(pb + 16);  // K 16..23 (or 24..31)
            }
            #pragma unroll
            for (int mi = 0; mi < 2; ++mi)
                #pragma unroll
                for (int ni = 0; ni < 4; ++ni)
                    acc[mi][ni] = __builtin_amdgcn_wmma_f32_16x16x32_bf16(
                        false, a[mi].v, false, bb[ni].v,
                        (short)0, acc[mi][ni], false, false);
        }
        __syncthreads();
    }

    // ---- epilogue: bias add + f32 store (C layout: M = r + 8*lhalf) ------
    float bias[4];
    #pragma unroll
    for (int ni = 0; ni < 4; ++ni)
        bias[ni] = b_out[nbase + waveN * 64 + ni * 16 + lmod];

    const size_t outBase = (size_t)bat * Tt * Uu * Vv;
    #pragma unroll
    for (int mi = 0; mi < 2; ++mi) {
        #pragma unroll
        for (int r = 0; r < 8; ++r) {
            int lrow = waveM * 32 + mi * 16 + lhalf * 8 + r;
            size_t m = (size_t)mblk * 128 + lrow;            // = t*64 + u
            float* op = out + outBase + m * Vv + nbase + waveN * 64 + lmod;
            #pragma unroll
            for (int ni = 0; ni < 4; ++ni)
                op[ni * 16] = acc[mi][ni][r] + bias[ni];
        }
    }
}

// ---------------------------------------------------------------------------
extern "C" void kernel_launch(void* const* d_in, const int* in_sizes, int n_in,
                              void* d_out, int out_size, void* d_ws, size_t ws_size,
                              hipStream_t stream)
{
    const float* encoder_out = (const float*)d_in[0];
    const float* decoder_out = (const float*)d_in[1];
    const float* W_enc  = (const float*)d_in[2];
    const float* b_enc  = (const float*)d_in[3];
    const float* g_enc  = (const float*)d_in[4];
    const float* be_enc = (const float*)d_in[5];
    const float* W_dec  = (const float*)d_in[6];
    const float* b_dec  = (const float*)d_in[7];
    const float* g_dec  = (const float*)d_in[8];
    const float* be_dec = (const float*)d_in[9];
    const float* W_out  = (const float*)d_in[10];
    const float* b_out  = (const float*)d_in[11];
    float* out = (float*)d_out;

    float* enc_ln = (float*)d_ws;                              // 4 MB
    float* dec_ln = enc_ln + (size_t)Bb * Tt * Jj;             // 0.5 MB
    unsigned short* Wt = (unsigned short*)(dec_ln + (size_t)Bb * Uu * Jj); // 1 MB

    proj_ln_kernel<<<dim3(Bb * Tt), dim3(256), 0, stream>>>(
        encoder_out, W_enc, b_enc, g_enc, be_enc, enc_ln);
    proj_ln_kernel<<<dim3(Bb * Uu), dim3(256), 0, stream>>>(
        decoder_out, W_dec, b_dec, g_dec, be_dec, dec_ln);
    wt_transpose_kernel<<<dim3(Vv / 64, Jj / 64), dim3(256), 0, stream>>>(
        W_out, Wt);

    joint_wmma_kernel<<<dim3(Vv / 128, (Tt * Uu) / 128, Bb), dim3(256), 0, stream>>>(
        enc_ln, dec_ln, Wt, b_out, out);
}
